// BiLSTM_29197187678853
// MI455X (gfx1250) — compile-verified, matrix operand order checked by reading, not run on hardware
//
#include <hip/hip_runtime.h>
#include <hip/hip_bf16.h>
#include <limits.h>

// ---------------------------------------------------------------------------
// Types matching the gfx1250 WMMA builtin signatures
// ---------------------------------------------------------------------------
typedef __attribute__((ext_vector_type(16))) __bf16 v16bf;
typedef __attribute__((ext_vector_type(8)))  __bf16 v8bf;
typedef __attribute__((ext_vector_type(8)))  float  v8f;

#define T_    256
#define P_    256
#define E_    300
#define EP_   320      // E padded to a multiple of 32 for WMMA K-steps
#define H_    256
#define H2_   512
#define H4_   1024
#define M_    50000
#define MPAD_ 50048    // padded to a multiple of 64 for the blocked GEMM
#define C_    16384
#define MT_   50000
#define CT_   16384

__device__ __forceinline__ float sigm(float x) { return 1.0f / (1.0f + expf(-x)); }

// order-preserving float->int transform for atomicMax-based segment max
__device__ __forceinline__ int orderF(float f) {
  int i = __float_as_int(f);
  return (i >= 0) ? i : (i ^ 0x7fffffff);
}
__device__ __forceinline__ float unorderF(int i) {
  return __int_as_float((i >= 0) ? i : (i ^ 0x7fffffff));
}

// ---------------------------------------------------------------------------
// WMMA fragment load. 16-bit A-matrix 16x32 layout (ISA 7.12.2):
//   lanes 0-15 : row M=lane,    K = 0..7  (VGPR0-3) and 16..23 (VGPR4-7)
//   lanes 16-31: row M=lane-16, K = 8..15 and 24..31
// i.e. per lane two contiguous 8-element (16B) chunks at +kb and +kb+16.
// B is stored pre-transposed (BT[N,K]) so the same loader works with lane=N.
// ---------------------------------------------------------------------------
__device__ __forceinline__ v16bf load_frag(const __bf16* __restrict__ p, int kb) {
  const v8bf lo = *(const v8bf*)(p + kb);
  const v8bf hi = *(const v8bf*)(p + kb + 16);
  return __builtin_shufflevector(lo, hi, 0, 1, 2, 3, 4, 5, 6, 7,
                                 8, 9, 10, 11, 12, 13, 14, 15);
}

__device__ __forceinline__ v8f wmma_bf16(v16bf a, v16bf b, v8f c) {
  return __builtin_amdgcn_wmma_f32_16x16x32_bf16(false, a, false, b,
                                                 (short)0, c, false, false);
}

// ---------------------------------------------------------------------------
// Blocked GEMM: C[M,N] (f32) = A[M,K](bf16) x B, B given as BT[N,K] (bf16).
// One wave computes a 64x32 tile (4x2 register block, 8 accumulators).
// Per k-step: 6 fragment loads (12x b128) feed 8 WMMAs. All fragments are
// loaded into distinct locals before any WMMA so the loads clause together
// and WMMA sources stay disjoint from next-iteration load targets.
// M multiple of 64, N multiple of 32, K multiple of 32.
// ---------------------------------------------------------------------------
__global__ void __launch_bounds__(256, 1)
gemm_bf16(const __bf16* __restrict__ A,
          const __bf16* __restrict__ BT,
          float* __restrict__ C,
          int Mblk, int Nblk, int K) {
  const int wave = blockIdx.x * (blockDim.x >> 5) + (threadIdx.x >> 5);
  if (wave >= Mblk * Nblk) return;                 // wave-uniform
  const int bm = wave / Nblk;
  const int bn = wave - bm * Nblk;
  const int lane = threadIdx.x & 31;
  const int r = lane & 15;
  const int kb = (lane >> 4) << 3;                 // 0 or 8
  const __bf16* pa0 = A + (size_t)(bm * 64 + r) * K + kb;
  const __bf16* pa1 = pa0 + (size_t)16 * K;
  const __bf16* pa2 = pa0 + (size_t)32 * K;
  const __bf16* pa3 = pa0 + (size_t)48 * K;
  const __bf16* pb0 = BT + (size_t)(bn * 32 + r) * K + kb;
  const __bf16* pb1 = pb0 + (size_t)16 * K;
  v8f acc00 = {}, acc01 = {}, acc10 = {}, acc11 = {};
  v8f acc20 = {}, acc21 = {}, acc30 = {}, acc31 = {};
  for (int kk = 0; kk < K; kk += 32) {
    const v16bf b0 = load_frag(pb0 + kk, 0);
    const v16bf b1 = load_frag(pb1 + kk, 0);
    const v16bf a0 = load_frag(pa0 + kk, 0);
    const v16bf a1 = load_frag(pa1 + kk, 0);
    const v16bf a2 = load_frag(pa2 + kk, 0);
    const v16bf a3 = load_frag(pa3 + kk, 0);
    acc00 = wmma_bf16(a0, b0, acc00);
    acc01 = wmma_bf16(a0, b1, acc01);
    acc10 = wmma_bf16(a1, b0, acc10);
    acc11 = wmma_bf16(a1, b1, acc11);
    acc20 = wmma_bf16(a2, b0, acc20);
    acc21 = wmma_bf16(a2, b1, acc21);
    acc30 = wmma_bf16(a3, b0, acc30);
    acc31 = wmma_bf16(a3, b1, acc31);
  }
  const int N = Nblk * 32;
  const int rbase = (lane < 16) ? 0 : 8;
  float* c0 = C + (size_t)(bm * 64 + rbase) * N + bn * 32 + r;
#pragma unroll
  for (int v = 0; v < 8; ++v) {
    c0[(size_t)v * N] = acc00[v];
    c0[(size_t)v * N + 16] = acc01[v];
    c0[(size_t)(16 + v) * N] = acc10[v];
    c0[(size_t)(16 + v) * N + 16] = acc11[v];
    c0[(size_t)(32 + v) * N] = acc20[v];
    c0[(size_t)(32 + v) * N + 16] = acc21[v];
    c0[(size_t)(48 + v) * N] = acc30[v];
    c0[(size_t)(48 + v) * N + 16] = acc31[v];
  }
}

// ---------------------------------------------------------------------------
// LSTM per-timestep GEMM: g[z] = x_t @ wih^T + h_z @ whh^T  (bias in pointwise)
// Latency-bound, tiny (256x1024, K=576): keep one 16x16 tile per wave so all
// 1024 waves/direction spread across the chip. blockIdx.z = direction.
// ---------------------------------------------------------------------------
__global__ void __launch_bounds__(256, 1)
k_lstm_gemm(const __bf16* __restrict__ xpad,
            const __bf16* __restrict__ wihTf,
            const __bf16* __restrict__ wihTb,
            const __bf16* __restrict__ hF,
            const __bf16* __restrict__ hB,
            const __bf16* __restrict__ whhTf,
            const __bf16* __restrict__ whhTb,
            float* __restrict__ g, int t) {
  const int z = blockIdx.z;
  const int tf = z ? (T_ - 1 - t) : t;
  const __bf16* A1 = xpad + (size_t)tf * P_ * EP_;
  const __bf16* B1 = z ? wihTb : wihTf;
  const __bf16* A2 = z ? hB : hF;
  const __bf16* B2 = z ? whhTb : whhTf;
  float* gz = g + (size_t)z * P_ * H4_;

  const int wave = blockIdx.x * (blockDim.x >> 5) + (threadIdx.x >> 5);
  const int tm = wave >> 6;        // Ntiles = 64
  const int tn = wave & 63;
  const int lane = threadIdx.x & 31;
  const int r = lane & 15;
  const int kb = (lane >> 4) << 3;
  v8f acc = {};
  {
    const __bf16* arow = A1 + (size_t)(tm * 16 + r) * EP_ + kb;
    const __bf16* brow = B1 + (size_t)(tn * 16 + r) * EP_ + kb;
#pragma unroll
    for (int kk = 0; kk < EP_; kk += 32) {
      const v16bf a = load_frag(arow + kk, 0);
      const v16bf b = load_frag(brow + kk, 0);
      acc = wmma_bf16(a, b, acc);
    }
  }
  {
    const __bf16* arow = A2 + (size_t)(tm * 16 + r) * H_ + kb;
    const __bf16* brow = B2 + (size_t)(tn * 16 + r) * H_ + kb;
#pragma unroll
    for (int kk = 0; kk < H_; kk += 32) {
      const v16bf a = load_frag(arow + kk, 0);
      const v16bf b = load_frag(brow + kk, 0);
      acc = wmma_bf16(a, b, acc);
    }
  }
  float* cptr = gz + (size_t)(tm * 16 + ((lane < 16) ? 0 : 8)) * H4_ + tn * 16 + r;
#pragma unroll
  for (int v = 0; v < 8; ++v) cptr[(size_t)v * H4_] = acc[v];
}

// LSTM pointwise: gates -> (c,h); writes bf16 h (next step) + bf16 embedding.
__global__ void k_lstm_pw(const float* __restrict__ g,
                          const float* __restrict__ b_f, const float* __restrict__ b_b,
                          float* __restrict__ cF, float* __restrict__ cB,
                          __bf16* __restrict__ hF, __bf16* __restrict__ hB,
                          __bf16* __restrict__ flat, int t) {
  const int z = blockIdx.z;
  const int idx = blockIdx.x * blockDim.x + threadIdx.x;   // 0..P_*H_-1
  const int p = idx >> 8;
  const int hh = idx & (H_ - 1);
  const int tf = z ? (T_ - 1 - t) : t;
  const float* gz = g + (size_t)z * P_ * H4_ + (size_t)p * H4_;
  const float* bias = z ? b_b : b_f;
  const float gi = gz[hh] + bias[hh];
  const float gf = gz[H_ + hh] + bias[H_ + hh];
  const float gg = gz[2 * H_ + hh] + bias[2 * H_ + hh];
  const float go = gz[3 * H_ + hh] + bias[3 * H_ + hh];
  float* cs = (z ? cB : cF) + idx;
  const float c = sigm(gf) * (*cs) + sigm(gi) * tanhf(gg);
  *cs = c;
  const float h = sigm(go) * tanhf(c);
  (z ? hB : hF)[idx] = (__bf16)h;
  flat[(size_t)(tf * P_ + p) * H2_ + z * H_ + hh] = (__bf16)h;  // [hf | hb]
}

__global__ void k_lstm_init(__bf16* hF, __bf16* hB, float* cF, float* cB) {
  int i = blockIdx.x * blockDim.x + threadIdx.x;
  if (i >= P_ * H_) return;
  hF[i] = (__bf16)0.0f; hB[i] = (__bf16)0.0f; cF[i] = 0.0f; cB[i] = 0.0f;
}

// ---------------------------------------------------------------------------
// Conversion kernels
// ---------------------------------------------------------------------------
// Row-major f32 -> bf16 with column zero-padding (also used as plain cvt).
__global__ void k_cvt_pad(const float* __restrict__ src, __bf16* __restrict__ dst,
                          long long rows, int cols, int colsPad) {
  long long i = (long long)blockIdx.x * blockDim.x + threadIdx.x;
  if (i >= rows * colsPad) return;
  long long rrow = i / colsPad;
  int col = (int)(i - rrow * colsPad);
  dst[i] = (col < cols) ? (__bf16)src[rrow * cols + col] : (__bf16)0.0f;
}

// w[K,N] f32 -> wt[N,K] bf16 (BT layout for WMMA B fragments)
__global__ void k_transpose_cvt(const float* __restrict__ w, __bf16* __restrict__ wt,
                                int K, int N) {
  long long i = (long long)blockIdx.x * blockDim.x + threadIdx.x;
  if (i >= (long long)K * N) return;
  int n = (int)(i / K);
  int k = (int)(i - (long long)n * K);
  wt[i] = (__bf16)w[(long long)k * N + n];
}

// ---------------------------------------------------------------------------
// Gathers (build bf16 A-matrices for the big WMMA GEMMs; zero-pad to rowsPad)
// ---------------------------------------------------------------------------
__global__ void k_gather2(const __bf16* __restrict__ flat, const int* __restrict__ o1,
                          const int* __restrict__ o2, __bf16* __restrict__ out,
                          int rows, int rowsPad) {
  long long i = (long long)blockIdx.x * blockDim.x + threadIdx.x;
  if (i >= (long long)rowsPad * H4_) return;
  int m = (int)(i >> 10);
  int col = (int)(i & 1023);
  if (m >= rows) { out[i] = (__bf16)0.0f; return; }
  int src = (col < H2_) ? o1[m] : o2[m];
  out[i] = flat[(size_t)src * H2_ + (col & (H2_ - 1))];
}

__global__ void k_gather3(const __bf16* __restrict__ flat, const int* __restrict__ o1,
                          const int* __restrict__ o2, const int* __restrict__ o3,
                          __bf16* __restrict__ out, int rows, int rowsPad) {
  long long i = (long long)blockIdx.x * blockDim.x + threadIdx.x;
  if (i >= (long long)rowsPad * 1536) return;
  int m = (int)(i / 1536);
  int col = (int)(i - (long long)m * 1536);
  if (m >= rows) { out[i] = (__bf16)0.0f; return; }
  const int* oc = (col < 512) ? o1 : ((col < 1024) ? o2 : o3);
  out[i] = flat[(size_t)oc[m] * H2_ + (col & (H2_ - 1))];
}

// feats = [pv2[idx0] | pv1[idx1] | pv0[idx2] | tv]  -> [CT, 2048] bf16
__global__ void k_gather_feats(const __bf16* __restrict__ pvB,   // [3,C,512]
                               const __bf16* __restrict__ tvB,   // [CT,512]
                               const int* __restrict__ tpi,      // [CT,3]
                               __bf16* __restrict__ out) {
  long long i = (long long)blockIdx.x * blockDim.x + threadIdx.x;
  if (i >= (long long)CT_ * 2048) return;
  int c = (int)(i >> 11);
  int col = (int)(i & 2047);
  int s = col >> 9;
  int d = col & 511;
  __bf16 v;
  if (s == 3) {
    v = tvB[(size_t)c * H2_ + d];
  } else {
    int ptype = 2 - s;
    int idx = tpi[c * 3 + s];
    v = pvB[((size_t)ptype * C_ + idx) * H2_ + d];
  }
  out[i] = v;
}

// ---------------------------------------------------------------------------
// Segment-max pooling (deterministic: int atomicMax on ordered float bits)
// ---------------------------------------------------------------------------
__global__ void k_pool_init(int* __restrict__ pooled, int* __restrict__ counts) {
  int i = blockIdx.x * blockDim.x + threadIdx.x;
  if (i < C_ * H2_) pooled[i] = INT_MIN;
  if (i < C_) counts[i] = 0;
}

__global__ void k_pool_scatter(const float* __restrict__ reps, const int* __restrict__ seg,
                               int* __restrict__ pooled, int* __restrict__ counts, int rows) {
  long long i = (long long)blockIdx.x * blockDim.x + threadIdx.x;
  if (i >= (long long)rows * H2_) return;
  int m = (int)(i >> 9);
  int d = (int)(i & 511);
  int s = seg[m];
  atomicMax(&pooled[(size_t)s * H2_ + d], orderF(reps[i]));
  if (d == 0) atomicAdd(&counts[s], 1);
}

// bias commutes with segment-max -> added here; empty segments use backoff.
__global__ void k_pool_final(const int* __restrict__ pooled, const int* __restrict__ counts,
                             const float* __restrict__ bias, const float* __restrict__ backoff,
                             float* __restrict__ vf, __bf16* __restrict__ vb) {
  int i = blockIdx.x * blockDim.x + threadIdx.x;
  if (i >= C_ * H2_) return;
  int c = i >> 9;
  int d = i & 511;
  float v = (counts[c] > 0) ? (unorderF(pooled[i]) + bias[d]) : backoff[d];
  v = tanhf(v);
  vf[i] = v;
  vb[i] = (__bf16)v;
}

// ---------------------------------------------------------------------------
// Logit heads
// ---------------------------------------------------------------------------
__global__ void k_pair_logits(const float* __restrict__ pv, const float* __restrict__ oW,
                              const float* __restrict__ ob, float* __restrict__ out) {
  int c = blockIdx.x * blockDim.x + threadIdx.x;
  if (c >= C_) return;
  const float* row = pv + (size_t)c * H2_;
  float s = 0.0f;
  for (int d = 0; d < H2_; ++d) s += row[d] * oW[d];
  out[c] = s + ob[0];
}

__global__ void k_triple_logits(const float* __restrict__ fin,  // [CT,512] pre-bias
                                const float* __restrict__ all_hb,
                                const float* __restrict__ out_tW,
                                const float* __restrict__ out_tb,
                                float* __restrict__ out) {
  int c = blockIdx.x * blockDim.x + threadIdx.x;
  if (c >= CT_) return;
  const float* row = fin + (size_t)c * H2_;
  float s = 0.0f;
  for (int d = 0; d < H2_; ++d) {
    float v = row[d] + all_hb[d];
    v = (v > 0.0f) ? v : 0.0f;
    s += v * out_tW[d];
  }
  out[c] = s + out_tb[0];
}

// ---------------------------------------------------------------------------
// Host orchestration (all launches on `stream`; graph-capture safe)
// ---------------------------------------------------------------------------
extern "C" void kernel_launch(void* const* d_in, const int* in_sizes, int n_in,
                              void* d_out, int out_size, void* d_ws, size_t ws_size,
                              hipStream_t stream) {
  (void)in_sizes; (void)n_in; (void)out_size; (void)ws_size;
  const float* x       = (const float*)d_in[0];
  const float* wih_f   = (const float*)d_in[1];
  const float* whh_f   = (const float*)d_in[2];
  const float* b_f     = (const float*)d_in[3];
  const float* wih_b   = (const float*)d_in[4];
  const float* whh_b   = (const float*)d_in[5];
  const float* b_b     = (const float*)d_in[6];
  const float* pair_hW = (const float*)d_in[7];
  const float* pair_hb = (const float*)d_in[8];
  const float* pair_oW = (const float*)d_in[9];
  const float* pair_ob = (const float*)d_in[10];
  const float* pair_bo = (const float*)d_in[11];
  const float* tri_hW  = (const float*)d_in[12];
  const float* tri_hb  = (const float*)d_in[13];
  const float* tri_bo  = (const float*)d_in[14];
  const float* all_hW  = (const float*)d_in[15];
  const float* all_hb  = (const float*)d_in[16];
  const float* out_tW  = (const float*)d_in[17];
  const float* out_tb  = (const float*)d_in[18];
  const int* occ1      = (const int*)d_in[19];
  const int* occ2      = (const int*)d_in[20];
  const int* seg       = (const int*)d_in[21];
  const int* tri_o1    = (const int*)d_in[22];
  const int* tri_o2    = (const int*)d_in[23];
  const int* tri_o3    = (const int*)d_in[24];
  const int* tri_seg   = (const int*)d_in[25];
  const int* tri_pi    = (const int*)d_in[26];
  float* out = (float*)d_out;   // [CT triple | 3*C pair]

  char* ws = (char*)d_ws;
  size_t off = 0;
  auto take = [&](size_t bytes) -> void* {
    void* p = ws + off;
    off = (off + bytes + 255) & ~(size_t)255;
    return p;
  };
  __bf16* xpad   = (__bf16*)take((size_t)T_ * P_ * EP_ * 2);
  __bf16* wihTf  = (__bf16*)take((size_t)H4_ * EP_ * 2);
  __bf16* wihTb  = (__bf16*)take((size_t)H4_ * EP_ * 2);
  __bf16* whhTf  = (__bf16*)take((size_t)H4_ * H_ * 2);
  __bf16* whhTb  = (__bf16*)take((size_t)H4_ * H_ * 2);
  __bf16* hF     = (__bf16*)take((size_t)P_ * H_ * 2);
  __bf16* hB     = (__bf16*)take((size_t)P_ * H_ * 2);
  float*  cF     = (float*) take((size_t)P_ * H_ * 4);
  float*  cB     = (float*) take((size_t)P_ * H_ * 4);
  float*  gbuf   = (float*) take((size_t)2 * P_ * H4_ * 4);
  __bf16* flat   = (__bf16*)take((size_t)T_ * P_ * H2_ * 2);
  __bf16* stage  = (__bf16*)take((size_t)MPAD_ * 1536 * 2);  // gather staging (max)
  float*  reps   = (float*) take((size_t)MPAD_ * H2_ * 4);   // reused as "final"
  int*    pooled = (int*)   take((size_t)C_ * H2_ * 4);
  int*    counts = (int*)   take((size_t)C_ * 4);
  float*  pvF    = (float*) take((size_t)3 * C_ * H2_ * 4);
  __bf16* pvB    = (__bf16*)take((size_t)3 * C_ * H2_ * 2);
  float*  tvF    = (float*) take((size_t)CT_ * H2_ * 4);
  __bf16* tvB    = (__bf16*)take((size_t)CT_ * H2_ * 2);
  __bf16* pairWT = (__bf16*)take((size_t)3 * H2_ * H4_ * 2);
  __bf16* triWT  = (__bf16*)take((size_t)H2_ * 1536 * 2);
  __bf16* allWT  = (__bf16*)take((size_t)H2_ * 2048 * 2);

  const int B = 256;  // 8 wave32 per block
  auto nb = [](long long n) { return (unsigned)((n + 255) / 256); };

  // --- weight / input conversions (bf16, BT layouts) ---
  k_cvt_pad<<<nb((long long)T_ * P_ * EP_), B, 0, stream>>>(x, xpad, (long long)T_ * P_, E_, EP_);
  k_cvt_pad<<<nb((long long)H4_ * EP_), B, 0, stream>>>(wih_f, wihTf, H4_, E_, EP_);
  k_cvt_pad<<<nb((long long)H4_ * EP_), B, 0, stream>>>(wih_b, wihTb, H4_, E_, EP_);
  k_cvt_pad<<<nb((long long)H4_ * H_), B, 0, stream>>>(whh_f, whhTf, H4_, H_, H_);
  k_cvt_pad<<<nb((long long)H4_ * H_), B, 0, stream>>>(whh_b, whhTb, H4_, H_, H_);
  for (int k = 0; k < 3; ++k)
    k_transpose_cvt<<<nb((long long)H4_ * H2_), B, 0, stream>>>(
        pair_hW + (size_t)k * H4_ * H2_, pairWT + (size_t)k * H2_ * H4_, H4_, H2_);
  k_transpose_cvt<<<nb((long long)1536 * H2_), B, 0, stream>>>(tri_hW, triWT, 1536, H2_);
  k_transpose_cvt<<<nb((long long)2048 * H2_), B, 0, stream>>>(all_hW, allWT, 2048, H2_);

  // --- BiLSTM (fwd + bwd interleaved via blockIdx.z) ---
  k_lstm_init<<<(P_ * H_ + B - 1) / B, B, 0, stream>>>(hF, hB, cF, cB);
  for (int t = 0; t < T_; ++t) {
    k_lstm_gemm<<<dim3(128, 1, 2), B, 0, stream>>>(xpad, wihTf, wihTb, hF, hB,
                                                   whhTf, whhTb, gbuf, t);
    k_lstm_pw<<<dim3(P_ * H_ / B, 1, 2), B, 0, stream>>>(gbuf, b_f, b_b, cF, cB,
                                                         hF, hB, flat, t);
  }

  // --- pair heads ---
  for (int k = 0; k < 3; ++k) {
    k_gather2<<<nb((long long)MPAD_ * H4_), B, 0, stream>>>(
        flat, occ1 + (size_t)k * M_, occ2 + (size_t)k * M_, stage, M_, MPAD_);
    gemm_bf16<<<((MPAD_ / 64) * (H2_ / 32) + 7) / 8, B, 0, stream>>>(
        stage, pairWT + (size_t)k * H2_ * H4_, reps, MPAD_ / 64, H2_ / 32, H4_);
    k_pool_init<<<(C_ * H2_ + B - 1) / B, B, 0, stream>>>(pooled, counts);
    k_pool_scatter<<<nb((long long)M_ * H2_), B, 0, stream>>>(
        reps, seg + (size_t)k * M_, pooled, counts, M_);
    k_pool_final<<<(C_ * H2_ + B - 1) / B, B, 0, stream>>>(
        pooled, counts, pair_hb + (size_t)k * H2_, pair_bo + (size_t)k * H2_,
        pvF + (size_t)k * C_ * H2_, pvB + (size_t)k * C_ * H2_);
    k_pair_logits<<<(C_ + B - 1) / B, B, 0, stream>>>(
        pvF + (size_t)k * C_ * H2_, pair_oW + (size_t)k * H2_, pair_ob + k,
        out + CT_ + (size_t)k * C_);
  }

  // --- triple head ---
  k_gather3<<<nb((long long)MPAD_ * 1536), B, 0, stream>>>(flat, tri_o1, tri_o2, tri_o3,
                                                           stage, MT_, MPAD_);
  gemm_bf16<<<((MPAD_ / 64) * (H2_ / 32) + 7) / 8, B, 0, stream>>>(
      stage, triWT, reps, MPAD_ / 64, H2_ / 32, 1536);
  k_pool_init<<<(C_ * H2_ + B - 1) / B, B, 0, stream>>>(pooled, counts);
  k_pool_scatter<<<nb((long long)MT_ * H2_), B, 0, stream>>>(reps, tri_seg, pooled,
                                                             counts, MT_);
  k_pool_final<<<(C_ * H2_ + B - 1) / B, B, 0, stream>>>(pooled, counts, tri_hb,
                                                         tri_bo, tvF, tvB);

  // --- final fused layer ---
  k_gather_feats<<<nb((long long)CT_ * 2048), B, 0, stream>>>(pvB, tvB, tri_pi, stage);
  gemm_bf16<<<((CT_ / 64) * (H2_ / 32) + 7) / 8, B, 0, stream>>>(
      stage, allWT, reps, CT_ / 64, H2_ / 32, 2048);
  k_triple_logits<<<(CT_ + B - 1) / B, B, 0, stream>>>(reps, all_hb, out_tW, out_tb, out);
}
